// DownSample_77996606095598
// MI455X (gfx1250) — matrix operand order repaired and checked by reading, not compiled, not run on hardware
//
#include <hip/hip_runtime.h>

// ---------------------------------------------------------------------------
// Sparse-conv inception stack for MI455X (gfx1250), wave32 + WMMA.
//
// Every layer is gather-GEMM: out[n,d] = sum_{k,c} src[nbr[n,k],c] * W[k,c,d].
// Implemented as 16x16x32 f16 WMMA tiles with f32 accumulation:
//   - activations stored as f16 (random gather traffic is the bottleneck;
//     the whole working set is L2-resident on the 192MB L2)
//   - weights prepacked once per layer to f16 W^T[d][kc], kc padded to 32,
//     then staged once per block into LDS (bank-conflict-free padded rows)
//   - A fragment gathered directly into the ISA register layout (no LDS
//     round trip), software-pipelined one K-chunk ahead
// ---------------------------------------------------------------------------

typedef __attribute__((ext_vector_type(16))) _Float16 v16h;
typedef __attribute__((ext_vector_type(8)))  _Float16 v8h;
typedef __attribute__((ext_vector_type(8)))  float    v8f;

union AB16 { v16h v; v8h g[2]; };

#define LDS_PADH 8   // 16B of pad per row -> conflict-free b128 row reads

// ---- weight prep: W[k][c][d] f32  ->  Wt[d][kc] f16, kc zero-padded to 32 --
__global__ void prep_weights(const float* __restrict__ W, _Float16* __restrict__ Wt,
                             int Ktot, int KtotPad, int Cout) {
  int i = blockIdx.x * blockDim.x + threadIdx.x;
  if (i >= KtotPad * Cout) return;
  int d  = i / KtotPad;
  int kc = i - d * KtotPad;
  float x = (kc < Ktot) ? W[(size_t)kc * Cout + d] : 0.0f;
  Wt[(size_t)d * KtotPad + kc] = (_Float16)x;
}

// ---- generic sparse-conv layer --------------------------------------------
// One wave per 16-voxel output tile; 4 waves per block share the LDS-staged
// weights. CIN/COUT are compile-time so the reduction-index division folds to
// a multiply and the accumulators are statically indexed VGPRs.
template<int CIN, int COUT, typename SrcT, typename OutT>
__global__ void __launch_bounds__(128)
sconv_wmma(const SrcT* __restrict__ src, const int* __restrict__ nbr, int Kk,
           int Ndst, const _Float16* __restrict__ Wt, int KtotPad,
           const float* __restrict__ bias,
           OutT* __restrict__ out, int out_stride, int col_off,
           const _Float16* __restrict__ res, int do_relu)
{
  constexpr int NT = COUT / 16;
  extern __shared__ char smem_raw[];
  _Float16* lds = (_Float16*)smem_raw;
  const int ldsStride = KtotPad + LDS_PADH;

  const int lane = threadIdx.x & 31;
  const int wave = threadIdx.x >> 5;
  const int tile = blockIdx.x * (blockDim.x >> 5) + wave;

  // ---- cooperative LDS staging of W^T (all waves, before any exit) --------
  // division-free 2D copy: each wave owns rows, each lane 16B chunks in-row
  // (32 lanes x 16B = 512B contiguous global reads per pass)
  for (int d = wave; d < COUT; d += 4) {
    const _Float16* gsrc = Wt + (size_t)d * KtotPad;
    _Float16*       ldst = lds + d * ldsStride;
    for (int kc = lane * 8; kc < KtotPad; kc += 32 * 8)
      *(v8h*)(ldst + kc) = *(const v8h*)(gsrc + kc);
  }
  __syncthreads();

  if (tile * 16 >= Ndst) return;          // wave-uniform: EXEC stays all-ones

  const int Ktot = Kk * CIN;
  const int m  = lane & 15;               // A row / B column / D column
  const int hi = lane >> 4;
  // 16-bit A 16x32 layout: lanes 0-15 hold K {0..7,16..23}, lanes 16-31 {8..15,24..31}
  const int a_base0 = hi * 8;
  const int a_base1 = 16 + hi * 8;
  // 16-bit B 32x16 layout (matches A): lanes 0-15 K {0..7,16..23}, lanes 16-31 {8..15,24..31}
  const int b_base  = hi * 8;

  int rowg = tile * 16 + m;               // gather row; clamp, stores guarded later
  if (rowg >= Ndst) rowg = Ndst - 1;
  if (nbr) __builtin_prefetch(nbr + (size_t)rowg * Kk, 0, 1);  // global_prefetch_b8

  // build one 16x32 A fragment by direct gather into the ISA register layout
  auto build_A = [&](int kc0) -> AB16 {
    AB16 a;
#pragma unroll
    for (int g = 0; g < 2; ++g) {
      const int kcb = kc0 + (g ? a_base1 : a_base0);
      v8h v = {};
      if constexpr ((CIN % 8) == 0 && sizeof(SrcT) == 2) {
        // 8 consecutive channels of one neighbor -> single aligned 16B load
        if (kcb < Ktot) {
          const int k   = kcb / CIN;
          const int c0  = kcb - k * CIN;
          const int idx = nbr ? nbr[rowg * Kk + k] : rowg;
          v = *(const v8h*)((const _Float16*)src + (size_t)idx * CIN + c0);
        }
      } else {
        // scalar path (Cin=3 f32 input conv)
#pragma unroll
        for (int e = 0; e < 8; ++e) {
          const int kc = kcb + e;
          float x = 0.0f;
          if (kc < Ktot) {
            const int k   = kc / CIN;           // CIN constant -> mul/shift
            const int c   = kc - k * CIN;
            const int idx = nbr ? nbr[rowg * Kk + k] : rowg;
            x = (float)src[(size_t)idx * CIN + c];
          }
          v[e] = (_Float16)x;
        }
      }
      a.g[g] = v;
    }
    return a;
  };

  v8f zacc = {};
  v8f acc[NT];
#pragma unroll
  for (int t = 0; t < NT; ++t) acc[t] = zacc;

  // ---- K loop, software-pipelined one chunk ahead -------------------------
  AB16 acur = build_A(0);
  for (int kc0 = 0; kc0 < KtotPad; kc0 += 32) {
    AB16 a = acur;
    const int kn = kc0 + 32;
    if (kn < KtotPad) acur = build_A(kn);       // loads overlap WMMAs below
#pragma unroll
    for (int t = 0; t < NT; ++t) {
      const _Float16* lw = lds + (t * 16 + m) * ldsStride + kc0 + b_base;
      AB16 b;
      b.g[0] = *(const v8h*)(lw);               // ds_load_b128
      b.g[1] = *(const v8h*)(lw + 16);
      acc[t] = __builtin_amdgcn_wmma_f32_16x16x32_f16(
          false, a.v, false, b.v, (short)0, acc[t], false, false);
    }
  }

  // -------- epilogue: D elem v of lane L -> M = v + hi*8, N = lane&15 ------
#pragma unroll
  for (int t = 0; t < NT; ++t) {
    const int n  = t * 16 + m;
    const float bv = bias[n];
#pragma unroll
    for (int v = 0; v < 8; ++v) {
      const int r = tile * 16 + v + hi * 8;
      if (r < Ndst) {
        float x = acc[t][v] + bv;
        if (res) x += (float)res[(size_t)r * 64 + col_off + n];  // residual stream is [*,64]
        if (do_relu && x < 0.0f) x = 0.0f;
        out[(size_t)r * out_stride + col_off + n] = (OutT)x;
      }
    }
  }
}

// ---------------------------------------------------------------------------
template<int CIN, int COUT, typename SrcT, typename OutT>
static void run_layer(const SrcT* src, const int* nbr, int Kk, int Ndst,
                      const _Float16* Wt, int KtotPad, const float* bias,
                      OutT* out, int out_stride, int col_off,
                      const _Float16* res, int do_relu, hipStream_t stream) {
  const int wpb    = 4;                       // 4 waves / 128 threads per block
  const int tiles  = (Ndst + 15) / 16;
  const int blocks = (tiles + wpb - 1) / wpb;
  const size_t smem = (size_t)(KtotPad + LDS_PADH) * 2 * COUT;   // <= 65KB (down0)
  sconv_wmma<CIN, COUT, SrcT, OutT><<<blocks, wpb * 32, smem, stream>>>(
      src, nbr, Kk, Ndst, Wt, KtotPad, bias, out, out_stride, col_off, res, do_relu);
}

static inline int find_sz(const int* s, int n, int sz, int occ) {
  for (int i = 0; i < n; ++i)
    if (s[i] == sz) { if (occ == 0) return i; --occ; }
  return 0;  // should not happen; keeps indexing safe
}

extern "C" void kernel_launch(void* const* d_in, const int* in_sizes, int n_in,
                              void* d_out, int out_size, void* d_ws, size_t ws_size,
                              hipStream_t stream) {
  constexpr int N_IN = 200000, N_OUT = 50000;

  // ---- locate inputs by flat element count (robust to pytree key order) ----
  const float* feats    = (const float*)d_in[find_sz(in_sizes, n_in, N_IN * 3, 0)];
  const int*   nbr_in3  = (const int*)  d_in[find_sz(in_sizes, n_in, N_IN * 27, 0)];
  const int*   nbr_down = (const int*)  d_in[find_sz(in_sizes, n_in, N_OUT * 8, 0)];
  const int*   nbr_out3 = (const int*)  d_in[find_sz(in_sizes, n_in, N_OUT * 27, 0)];
  const float* conv0_w  = (const float*)d_in[find_sz(in_sizes, n_in, 27 * 3 * 64, 0)];
  const float* down0_w  = (const float*)d_in[find_sz(in_sizes, n_in, 8 * 3 * 64, 0)];
  const float* conv0_b  = (const float*)d_in[find_sz(in_sizes, n_in, 64, 0)];
  const float* down0_b  = (const float*)d_in[find_sz(in_sizes, n_in, 64, 1)];

  struct Blk { const float *w00,*b00,*w01,*b01,*w10,*b10,*w11,*b11,*w12,*b12; } blk[3];
  for (int b = 0; b < 3; ++b) {
    int i = find_sz(in_sizes, n_in, 27 * 64 * 16, b);   // w00 of block b
    auto P = [&](int j){ return (const float*)d_in[j]; };
    if (i + 1 < n_in && in_sizes[i + 1] == 27 * 16 * 32) {
      // sorted-key pytree: [b00,b01,b10,b11,b12,w00,w01,w10,w11,w12]
      blk[b] = { P(i), P(i-5), P(i+1), P(i-4), P(i+2), P(i-3), P(i+3), P(i-2), P(i+4), P(i-1) };
    } else {
      // insertion order: [w00,b00,w01,b01,w10,b10,w11,b11,w12,b12]
      blk[b] = { P(i), P(i+1), P(i+2), P(i+3), P(i+4), P(i+5), P(i+6), P(i+7), P(i+8), P(i+9) };
    }
  }

  // ---- carve workspace (harness poisons it; every byte we read is written first)
  char* wsp = (char*)d_ws;
  auto carve = [&](size_t bytes) { char* p = wsp; wsp += (bytes + 255) & ~(size_t)255; return p; };
  _Float16* h0  = (_Float16*)carve((size_t)N_IN  * 64 * 2);
  _Float16* hA  = (_Float16*)carve((size_t)N_OUT * 64 * 2);
  _Float16* hB  = (_Float16*)carve((size_t)N_OUT * 64 * 2);
  _Float16* t00 = (_Float16*)carve((size_t)N_OUT * 16 * 2);
  _Float16* t10 = (_Float16*)carve((size_t)N_OUT * 16 * 2);
  _Float16* t11 = (_Float16*)carve((size_t)N_OUT * 16 * 2);
  _Float16* wt_conv0 = (_Float16*)carve((size_t)64 * 96   * 2);
  _Float16* wt_down0 = (_Float16*)carve((size_t)64 * 512  * 2);
  _Float16* wt_w00   = (_Float16*)carve((size_t)16 * 1728 * 2);
  _Float16* wt_w01   = (_Float16*)carve((size_t)32 * 448  * 2);
  _Float16* wt_w10   = (_Float16*)carve((size_t)16 * 64   * 2);
  _Float16* wt_w11   = (_Float16*)carve((size_t)16 * 448  * 2);
  _Float16* wt_w12   = (_Float16*)carve((size_t)32 * 32   * 2);
  (void)ws_size; (void)out_size;

  auto prep = [&](const float* W, _Float16* Wt, int Ktot, int KtotPad, int Cout) {
    int total = KtotPad * Cout;
    prep_weights<<<(total + 255) / 256, 256, 0, stream>>>(W, Wt, Ktot, KtotPad, Cout);
  };

  // ---- stem ----------------------------------------------------------------
  prep(conv0_w, wt_conv0, 81, 96, 64);
  prep(down0_w, wt_down0, 512, 512, 64);
  run_layer<3, 64, float, _Float16>(feats, nbr_in3, 27, N_IN,
      wt_conv0, 96, conv0_b, h0, 64, 0, nullptr, 1, stream);
  run_layer<64, 64, _Float16, _Float16>(h0, nbr_down, 8, N_OUT,
      wt_down0, 512, down0_b, hA, 64, 0, nullptr, 1, stream);

  // ---- inception blocks (stream-ordered weight slot reuse) -----------------
  _Float16* hcur = hA;
  _Float16* hnext = hB;
  for (int b = 0; b < 3; ++b) {
    prep(blk[b].w00, wt_w00, 1728, 1728, 16);
    prep(blk[b].w01, wt_w01,  432,  448, 32);
    prep(blk[b].w10, wt_w10,   64,   64, 16);
    prep(blk[b].w11, wt_w11,  432,  448, 16);
    prep(blk[b].w12, wt_w12,   16,   32, 32);

    // branch 0: conv3 C->C/4, relu
    run_layer<64, 16, _Float16, _Float16>(hcur, nbr_out3, 27, N_OUT,
        wt_w00, 1728, blk[b].b00, t00, 16, 0, nullptr, 1, stream);
    // branch 1: 1x1 C->C/4 relu ; conv3 C/4->C/4 relu
    run_layer<64, 16, _Float16, _Float16>(hcur, nullptr, 1, N_OUT,
        wt_w10, 64, blk[b].b10, t10, 16, 0, nullptr, 1, stream);
    run_layer<16, 16, _Float16, _Float16>(t10, nbr_out3, 27, N_OUT,
        wt_w11, 448, blk[b].b11, t11, 16, 0, nullptr, 1, stream);

    const bool last = (b == 2);
    if (!last) {
      // concat(out0,out1)+h  ==  column-offset writes with fused residual add
      run_layer<16, 32, _Float16, _Float16>(t00, nbr_out3, 27, N_OUT,
          wt_w01, 448, blk[b].b01, hnext, 64, 0, hcur, 0, stream);
      run_layer<16, 32, _Float16, _Float16>(t11, nullptr, 1, N_OUT,
          wt_w12, 32, blk[b].b12, hnext, 64, 32, hcur, 0, stream);
      _Float16* t = hcur; hcur = hnext; hnext = t;
    } else {
      // final block writes f32 straight into d_out
      run_layer<16, 32, _Float16, float>(t00, nbr_out3, 27, N_OUT,
          wt_w01, 448, blk[b].b01, (float*)d_out, 64, 0, hcur, 0, stream);
      run_layer<16, 32, _Float16, float>(t11, nullptr, 1, N_OUT,
          wt_w12, 32, blk[b].b12, (float*)d_out, 64, 32, hcur, 0, stream);
    }
  }
}
